// TripletAggregateUngated_71012989272670
// MI455X (gfx1250) — compile-verified, matrix operand order checked by reading, not run on hardware
//
#include <hip/hip_runtime.h>
#include <hip/hip_bf16.h>

// ---------------- problem constants ----------------
#define BB 2
#define NN 256
#define WW 128
#define HH 8
#define DD 16
#define MROWS (BB * NN * NN)       // 131072 edge rows

typedef __attribute__((ext_vector_type(16))) __bf16 v16bf;
typedef __attribute__((ext_vector_type(8)))  __bf16 v8bf;
typedef __attribute__((ext_vector_type(8)))  float  v8f;

// ---------------------------------------------------------------------------
// Wave-level 32x32 output macro-tile GEMM (2x2 blocking of 16x16 WMMA tiles).
//   C[t][u] += A[16t..16t+15, :] * B[:, 16u..16u+15]
// A: row-major [32 x K], leading dim lda.
// Bt: TRANSPOSED B, row-major [32 x K] (row n holds column n of B), ld ldbt.
// Fragment layouts per CDNA5 ISA 7.12.2 (bf16 16x32 A / 32x16 B):
//   lane&15 = M (or N); lane>>4 selects K-halves {kg*8..kg*8+7, 16+kg*8..}
// All fragment loads are two contiguous 16-byte b128 loads per operand.
// ---------------------------------------------------------------------------
template <int K>
__device__ inline void wmma_gemm_2x2(const __bf16* __restrict__ A, int lda,
                                     const __bf16* __restrict__ Bt, int ldbt,
                                     v8f acc[2][2]) {
  const int lane = threadIdx.x & 31;
  const int r    = lane & 15;
  const int kg   = (lane >> 4) & 1;
  const __bf16* arow0 = A  + (size_t)r * lda;
  const __bf16* arow1 = A  + (size_t)(r + 16) * lda;
  const __bf16* brow0 = Bt + (size_t)r * ldbt;
  const __bf16* brow1 = Bt + (size_t)(r + 16) * ldbt;
#pragma unroll
  for (int k0 = 0; k0 < K; k0 += 32) {
    const int o0 = k0 + kg * 8;        // first K-half chunk
    const int o1 = o0 + 16;            // second K-half chunk
    v8bf a0l = *(const v8bf*)(arow0 + o0), a0h = *(const v8bf*)(arow0 + o1);
    v8bf a1l = *(const v8bf*)(arow1 + o0), a1h = *(const v8bf*)(arow1 + o1);
    v8bf b0l = *(const v8bf*)(brow0 + o0), b0h = *(const v8bf*)(brow0 + o1);
    v8bf b1l = *(const v8bf*)(brow1 + o0), b1h = *(const v8bf*)(brow1 + o1);
    v16bf a0 = __builtin_shufflevector(a0l, a0h, 0,1,2,3,4,5,6,7,8,9,10,11,12,13,14,15);
    v16bf a1 = __builtin_shufflevector(a1l, a1h, 0,1,2,3,4,5,6,7,8,9,10,11,12,13,14,15);
    v16bf b0 = __builtin_shufflevector(b0l, b0h, 0,1,2,3,4,5,6,7,8,9,10,11,12,13,14,15);
    v16bf b1 = __builtin_shufflevector(b1l, b1h, 0,1,2,3,4,5,6,7,8,9,10,11,12,13,14,15);
    acc[0][0] = __builtin_amdgcn_wmma_f32_16x16x32_bf16(false, a0, false, b0, (short)0, acc[0][0], false, false);
    acc[0][1] = __builtin_amdgcn_wmma_f32_16x16x32_bf16(false, a0, false, b1, (short)0, acc[0][1], false, false);
    acc[1][0] = __builtin_amdgcn_wmma_f32_16x16x32_bf16(false, a1, false, b0, (short)0, acc[1][0], false, false);
    acc[1][1] = __builtin_amdgcn_wmma_f32_16x16x32_bf16(false, a1, false, b1, (short)0, acc[1][1], false, false);
  }
}

// ---------------------------------------------------------------------------
// Kernel 0: convert + TRANSPOSE weights to bf16.
//   WvT [2W][W] : WvT[n][k] = W_V[k][n]
//   WoT [W][2W] : WoT[n][k] = W_O[k][n]
// ---------------------------------------------------------------------------
__global__ void k_cvt_weights(const float* __restrict__ Wv,
                              const float* __restrict__ Wo,
                              __bf16* __restrict__ WvT,
                              __bf16* __restrict__ WoT) {
  int t = blockIdx.x * blockDim.x + threadIdx.x;
  if (t < WW * 2 * WW) {  // 32768
    int k = t / (2 * WW), n = t % (2 * WW);     // Wv is [WW][2WW]
    WvT[(size_t)n * WW + k] = (__bf16)Wv[t];
    int k2 = t / WW, n2 = t % WW;               // Wo is [2WW][WW]
    WoT[(size_t)n2 * (2 * WW) + k2] = (__bf16)Wo[t];
  }
}

// ---------------------------------------------------------------------------
// Kernel 1: LayerNorm over W=128 + E projection (128 -> 16).
// One block (128 threads) per edge row. Writes e_ln (bf16) and E (fp32).
// ---------------------------------------------------------------------------
__global__ void k_ln_eproj(const float* __restrict__ e,
                           const float* __restrict__ ln_g,
                           const float* __restrict__ ln_b,
                           const float* __restrict__ W_E,
                           const float* __restrict__ b_E,
                           __bf16* __restrict__ eln,
                           float*  __restrict__ Eout) {
  const int row = blockIdx.x;        // [0, MROWS)
  const int t   = threadIdx.x;       // 128
  __shared__ float red[WW];
  __shared__ float y[WW];

  const float x = e[(size_t)row * WW + t];

  red[t] = x; __syncthreads();
  for (int s = 64; s > 0; s >>= 1) { if (t < s) red[t] += red[t + s]; __syncthreads(); }
  const float mu = red[0] * (1.0f / WW);
  __syncthreads();

  const float dx = x - mu;
  red[t] = dx * dx; __syncthreads();
  for (int s = 64; s > 0; s >>= 1) { if (t < s) red[t] += red[t + s]; __syncthreads(); }
  const float var = red[0] * (1.0f / WW);

  const float yn = dx * __frsqrt_rn(var + 1e-5f) * ln_g[t] + ln_b[t];
  y[t] = yn;
  eln[(size_t)row * WW + t] = (__bf16)yn;
  __syncthreads();

  // E[row][o] = sum_c y[c] * W_E[c][o] + b_E[o],  o in [0,16)
  const int o = t & 15, seg = t >> 4;            // seg covers c in [seg*16, seg*16+16)
  float p = 0.0f;
#pragma unroll
  for (int cc = 0; cc < 16; ++cc) p += y[seg * 16 + cc] * W_E[(seg * 16 + cc) * 16 + o];
  red[t] = p; __syncthreads();
  if (t < 16) {
    float s = 0.0f;
#pragma unroll
    for (int g = 0; g < 8; ++g) s += red[g * 16 + t];
    Eout[(size_t)row * 16 + t] = s + b_E[t];
  }
}

// ---------------------------------------------------------------------------
// Kernel 2: A_in = softmax_k(E_in[b,i,k,h] + mask[b,i,k]).
// Block = (b,i), thread = k. Stores bf16 at [b*H+h][i][k] (row-major 256x256).
// ---------------------------------------------------------------------------
__global__ void k_softmax_in(const float* __restrict__ E,
                             const float* __restrict__ mask,
                             __bf16* __restrict__ Ain) {
  const int bi = blockIdx.x;          // b*N + i
  const int b  = bi >> 8;
  const int i  = bi & 255;
  const int k  = threadIdx.x;         // 256
  __shared__ float red[NN];

  const float m0 = mask[(size_t)bi * NN + k];
  float v[HH];
  const float* src = E + ((size_t)bi * NN + k) * 16;
#pragma unroll
  for (int h = 0; h < HH; ++h) v[h] = src[h] + m0;

  for (int h = 0; h < HH; ++h) {
    red[k] = v[h]; __syncthreads();
    for (int s = 128; s > 0; s >>= 1) { if (k < s) red[k] = fmaxf(red[k], red[k + s]); __syncthreads(); }
    const float mx = red[0]; __syncthreads();
    const float ex = __expf(v[h] - mx);
    red[k] = ex; __syncthreads();
    for (int s = 128; s > 0; s >>= 1) { if (k < s) red[k] += red[k + s]; __syncthreads(); }
    const float sum = red[0]; __syncthreads();
    Ain[(((size_t)(b * HH + h) * NN + i) * NN) + k] = (__bf16)(ex / sum);
  }
}

// ---------------------------------------------------------------------------
// Kernel 3: A_out softmax over axis=1 (k of E_out[b,k,i,h] + mask[b,k,i]).
// Stores TRANSPOSED: [b*H+h][i][k] = A_out[b,k,i,h].
// ---------------------------------------------------------------------------
__global__ void k_softmax_out(const float* __restrict__ E,
                              const float* __restrict__ mask,
                              __bf16* __restrict__ AoutT) {
  const int bi = blockIdx.x;
  const int b  = bi >> 8;
  const int i  = bi & 255;
  const int k  = threadIdx.x;
  __shared__ float red[NN];

  const float m0 = mask[((size_t)(b * NN + k) * NN) + i];
  float v[HH];
  const float* src = E + ((size_t)(b * NN + k) * NN + i) * 16 + 8;
#pragma unroll
  for (int h = 0; h < HH; ++h) v[h] = src[h] + m0;

  for (int h = 0; h < HH; ++h) {
    red[k] = v[h]; __syncthreads();
    for (int s = 128; s > 0; s >>= 1) { if (k < s) red[k] = fmaxf(red[k], red[k + s]); __syncthreads(); }
    const float mx = red[0]; __syncthreads();
    const float ex = __expf(v[h] - mx);
    red[k] = ex; __syncthreads();
    for (int s = 128; s > 0; s >>= 1) { if (k < s) red[k] += red[k + s]; __syncthreads(); }
    const float sum = red[0]; __syncthreads();
    AoutT[(((size_t)(b * HH + h) * NN + i) * NN) + k] = (__bf16)(ex / sum);
  }
}

// ---------------------------------------------------------------------------
// Kernel 4: V projection [131072x128] @ [128x256] via WMMA (2x2 macro tiles),
// scattering into TRANSPOSED per-(b,h) aggregation operands:
//   VinT [b,h][r1*16+d][r2] = V[b,r1,r2, d*8+h]        (c <  128)
//   VoutT[b,h][r2*16+d][r1] = V[b,r1,r2, 128+d*8+h]    (c >= 128)
// Block = 8 waves; wave = one 32-row macro tile; grid.y = 32-col macro tile.
// ---------------------------------------------------------------------------
__global__ void k_vproj(const __bf16* __restrict__ eln,
                        const __bf16* __restrict__ WvT,
                        const float*  __restrict__ bV,
                        __bf16* __restrict__ VinT,
                        __bf16* __restrict__ VoutT) {
  const int wave = threadIdx.x >> 5;
  const int lane = threadIdx.x & 31;
  const int m0   = (blockIdx.x * 8 + wave) * 32;  // row macro-tile base
  const int n0   = blockIdx.y * 32;               // col macro-tile base

  v8f acc[2][2] = {};
  wmma_gemm_2x2<WW>(eln + (size_t)m0 * WW, WW, WvT + (size_t)n0 * WW, WW, acc);

  const int rhalf = 8 * (lane >> 4);
#pragma unroll
  for (int u = 0; u < 2; ++u) {
    const int col  = n0 + u * 16 + (lane & 15);
    const float bias = bV[col];
#pragma unroll
    for (int t = 0; t < 2; ++t) {
#pragma unroll
      for (int i2 = 0; i2 < 8; ++i2) {
        const int gr  = m0 + t * 16 + i2 + rhalf;   // global row in [0, MROWS)
        const int b   = gr >> 16;                   // / (N*N)
        const int rem = gr & 65535;
        const int r1  = rem >> 8, r2 = rem & 255;
        const float val = acc[t][u][i2] + bias;
        if (col < WW) {
          const int d = col >> 3, h = col & 7;
          VinT[((size_t)(b * HH + h) * (NN * DD) + r1 * DD + d) * NN + r2] = (__bf16)val;
        } else {
          const int c2 = col - WW;
          const int d = c2 >> 3, h = c2 & 7;
          VoutT[((size_t)(b * HH + h) * (NN * DD) + r2 * DD + d) * NN + r1] = (__bf16)val;
        }
      }
    }
  }
}

// ---------------------------------------------------------------------------
// Kernel 5: aggregation GEMMs. Per (b,h, in/out):
//   C[i, jd] = sum_k A[i,k] * VT[jd][k]    (256x256 @ (4096x256)^T)
// grid.z in [0,32): z<16 -> "in" slice bh=z, else "out" slice bh=z-16.
// Writes Va bf16 [B,N,N,256] with channel = d*16 + h (+8 for out).
// ---------------------------------------------------------------------------
__global__ void k_attn(const __bf16* __restrict__ Ain,
                       const __bf16* __restrict__ AoutT,
                       const __bf16* __restrict__ VinT,
                       const __bf16* __restrict__ VoutT,
                       __bf16* __restrict__ Va) {
  const int z      = blockIdx.z;
  const bool isout = z >= (BB * HH);
  const int bh     = isout ? z - BB * HH : z;

  const __bf16* Ab  = (isout ? AoutT : Ain) + (size_t)bh * NN * NN;
  const __bf16* VbT = (isout ? VoutT : VinT) + (size_t)bh * (NN * DD) * NN;

  const int wave = threadIdx.x >> 5;
  const int lane = threadIdx.x & 31;
  const int i0   = blockIdx.y * 32;                        // row macro tile
  const int n0   = (blockIdx.x * 8 + wave) * 32;           // col macro tile in [0,4096)

  v8f acc[2][2] = {};
  wmma_gemm_2x2<NN>(Ab + (size_t)i0 * NN, NN, VbT + (size_t)n0 * NN, NN, acc);

  const int b  = bh >> 3, h = bh & 7;
  const int rhalf = 8 * (lane >> 4);
#pragma unroll
  for (int u = 0; u < 2; ++u) {
    const int jd = n0 + u * 16 + (lane & 15);
    const int j  = jd >> 4, d = jd & 15;
    const int ch = d * 16 + h + (isout ? 8 : 0);
#pragma unroll
    for (int t = 0; t < 2; ++t) {
#pragma unroll
      for (int i2 = 0; i2 < 8; ++i2) {
        const int i = i0 + t * 16 + i2 + rhalf;
        Va[(((size_t)(b * NN + i) * NN + j) * (2 * WW)) + ch] = (__bf16)acc[t][u][i2];
      }
    }
  }
}

// ---------------------------------------------------------------------------
// Kernel 6: output projection [131072x256] @ [256x128] + bias, fp32 out.
// ---------------------------------------------------------------------------
__global__ void k_oproj(const __bf16* __restrict__ Va,
                        const __bf16* __restrict__ WoT,
                        const float*  __restrict__ bO,
                        float* __restrict__ out) {
  const int wave = threadIdx.x >> 5;
  const int lane = threadIdx.x & 31;
  const int m0   = (blockIdx.x * 8 + wave) * 32;
  const int n0   = blockIdx.y * 32;

  v8f acc[2][2] = {};
  wmma_gemm_2x2<2 * WW>(Va + (size_t)m0 * (2 * WW), 2 * WW,
                        WoT + (size_t)n0 * (2 * WW), 2 * WW, acc);

  const int rhalf = 8 * (lane >> 4);
#pragma unroll
  for (int u = 0; u < 2; ++u) {
    const int col  = n0 + u * 16 + (lane & 15);
    const float bias = bO[col];
#pragma unroll
    for (int t = 0; t < 2; ++t) {
#pragma unroll
      for (int i2 = 0; i2 < 8; ++i2) {
        out[(size_t)(m0 + t * 16 + i2 + rhalf) * WW + col] = acc[t][u][i2] + bias;
      }
    }
  }
}

// ---------------------------------------------------------------------------
extern "C" void kernel_launch(void* const* d_in, const int* in_sizes, int n_in,
                              void* d_out, int out_size, void* d_ws, size_t ws_size,
                              hipStream_t stream) {
  const float* e    = (const float*)d_in[0];
  const float* mask = (const float*)d_in[1];
  const float* ln_g = (const float*)d_in[2];
  const float* ln_b = (const float*)d_in[3];
  const float* W_V  = (const float*)d_in[4];
  const float* b_V  = (const float*)d_in[5];
  const float* W_E  = (const float*)d_in[6];
  const float* b_E  = (const float*)d_in[7];
  const float* W_O  = (const float*)d_in[8];
  const float* b_O  = (const float*)d_in[9];
  float* out = (float*)d_out;

  char* ws = (char*)d_ws;
  size_t off = 0;
  auto take = [&](size_t bytes) -> char* {
    char* p = ws + off;
    off = (off + bytes + 255) & ~(size_t)255;
    return p;
  };

  __bf16* eln   = (__bf16*)take((size_t)MROWS * WW * sizeof(__bf16));             // 33.5 MB
  float*  Ebuf  = (float*) take((size_t)MROWS * 16 * sizeof(float));              //  8.4 MB
  __bf16* Ain   = (__bf16*)take((size_t)BB * HH * NN * NN * sizeof(__bf16));      //  2.1 MB
  __bf16* AoutT = (__bf16*)take((size_t)BB * HH * NN * NN * sizeof(__bf16));      //  2.1 MB
  __bf16* VinT  = (__bf16*)take((size_t)BB * HH * NN * NN * DD * sizeof(__bf16)); // 33.5 MB
  __bf16* VoutT = (__bf16*)take((size_t)BB * HH * NN * NN * DD * sizeof(__bf16)); // 33.5 MB
  __bf16* Vab   = (__bf16*)take((size_t)MROWS * 2 * WW * sizeof(__bf16));         // 67 MB
  __bf16* WvT   = (__bf16*)take((size_t)WW * 2 * WW * sizeof(__bf16));
  __bf16* WoT   = (__bf16*)take((size_t)2 * WW * WW * sizeof(__bf16));

  k_cvt_weights<<<128, 256, 0, stream>>>(W_V, W_O, WvT, WoT);
  k_ln_eproj<<<MROWS, WW, 0, stream>>>(e, ln_g, ln_b, W_E, b_E, eln, Ebuf);
  k_softmax_in <<<BB * NN, NN, 0, stream>>>(Ebuf, mask, Ain);
  k_softmax_out<<<BB * NN, NN, 0, stream>>>(Ebuf, mask, AoutT);
  k_vproj<<<dim3(MROWS / 32 / 8, 8), 256, 0, stream>>>(eln, WvT, b_V, VinT, VoutT);
  k_attn <<<dim3(16, 8, 2 * BB * HH), 256, 0, stream>>>(Ain, AoutT, VinT, VoutT, Vab);
  k_oproj<<<dim3(MROWS / 32 / 8, 4), 256, 0, stream>>>(Vab, WoT, b_O, out);
}
